// DeformableConv2d_7541962571941
// MI455X (gfx1250) — matrix operand order repaired and checked by reading, not compiled, not run on hardware
//
#include <hip/hip_runtime.h>
#include <hip/hip_bf16.h>
#include <math.h>

// ---------------------------------------------------------------------------
// Deformable Conv v2 for MI455X (gfx1250, wave32, WMMA)
//   K1: offset/mask convs as implicit GEMM:  M=32(27 pad) x K=576 x N=16384/b
//   K2: fused bilinear-sample implicit GEMM: M=64        x K=576 x N=16384/b
//   f16 A/B fragments, f32 accum, double-buffered B tile, branchless gathers
//   batched ahead of a single packed ds_store_b128 per thread per chunk.
// ---------------------------------------------------------------------------

typedef __attribute__((ext_vector_type(16))) _Float16 v16h;
typedef __attribute__((ext_vector_type(8)))  _Float16 v8h;
typedef __attribute__((ext_vector_type(4)))  _Float16 v4h;
typedef __attribute__((ext_vector_type(8)))  float    v8f;

constexpr int B_   = 4;
constexpr int C_   = 64;
constexpr int H_   = 128;
constexpr int W_   = 128;
constexpr int CO_  = 64;
constexpr int KPOS = 9;                  // 3x3
constexpr int CK   = C_ * KPOS;          // 576 (GEMM K)
constexpr int HW   = H_ * W_;            // 16384
constexpr int OFF_CH = 18, MOD_CH = 9;

constexpr int NT      = 128;             // pixel tile (GEMM N per block)
constexpr int KC      = 32;              // WMMA K chunk
constexpr int NCHUNK  = CK / KC;         // 18
constexpr int BST     = 136;             // B-tile row stride (f16): 272B, 16B-aligned
constexpr int NTHREADS = 512;            // 16 waves

union Frag { v16h v; v8h h[2]; };

// ===========================================================================
// Kernel 1: offset(18)+mask(9) 3x3 convs as WMMA implicit GEMM (M padded to 32)
// ===========================================================================
constexpr size_t K1_A_BYTES  = (size_t)32 * CK * 2;           // 36,864
constexpr size_t K1_BT_BYTES = (size_t)2 * KC * BST * 2;      // 17,408
constexpr size_t K1_SMEM     = K1_A_BYTES + K1_BT_BYTES;      // 54,272

__global__ void __launch_bounds__(NTHREADS)
conv_offmask_wmma(const float* __restrict__ x,
                  const float* __restrict__ w_off, const float* __restrict__ b_off,
                  const float* __restrict__ w_mod, const float* __restrict__ b_mod,
                  float* __restrict__ offset, float* __restrict__ mask)
{
    extern __shared__ __align__(16) char smem[];
    _Float16* a_lds  = (_Float16*)smem;                       // [32][576] f16
    _Float16* b_tile = (_Float16*)(smem + K1_A_BYTES);        // [2][32][BST] f16

    const int tid = threadIdx.x;
    const int bid = blockIdx.x;
    const int b       = bid >> 7;                 // 128 tiles per batch
    const int pixbase = (bid & 127) * NT;

    // A = [w_off ; w_mod ; zero-pad] rows, f16 (float4 -> packed 4xf16 stores)
    for (int i = tid; i < 32 * CK / 4; i += NTHREADS) {
        int e  = i * 4;
        int oc = e / CK, rem = e - oc * CK;
        float4 v;
        if (oc < OFF_CH)      v = *(const float4*)&w_off[oc * CK + rem];
        else if (oc < 27)     v = *(const float4*)&w_mod[(oc - OFF_CH) * CK + rem];
        else                  v = make_float4(0.f, 0.f, 0.f, 0.f);
        v4h h4; h4[0] = (_Float16)v.x; h4[1] = (_Float16)v.y;
                h4[2] = (_Float16)v.z; h4[3] = (_Float16)v.w;
        *(v4h*)&a_lds[e] = h4;
    }

    const float* xb = x + (size_t)b * C_ * HW;
    const int lane  = tid & 31;
    const int wave  = tid >> 5;                   // 16 waves
    const int mtile = wave & 1;                   // 2 M-tiles of 16
    const int ntile = wave >> 1;                  // 8 N-tiles of 16
    const int mrow  = mtile * 16 + (lane & 15);
    const int khalf = lane >> 4;

    // per-thread B-tile slot: 8 consecutive pixels of one K-row (same output row)
    const int bkrow = (tid * 8) >> 7;
    const int bpix0 = (tid * 8) & 127;
    const int ho    = (pixbase + bpix0) >> 7;     // constant per thread
    const int wo0   = bpix0;                      // pixbase row-aligned

    auto build = [&](int ch, int buf) {
        int kt = ch * KC + bkrow;
        int c  = kt / KPOS, kp = kt - c * KPOS;
        int dh = kp / 3 - 1, dw = kp % 3 - 1;
        int h  = ho + dh;
        int hc = h < 0 ? 0 : (h > H_ - 1 ? H_ - 1 : h);
        float rowv = (h >= 0 && h < H_) ? 1.f : 0.f;
        const float* rp = xb + (size_t)c * HW + hc * W_;
        float g[8], vm[8];
        #pragma unroll
        for (int i = 0; i < 8; ++i) {             // branchless: clamp + always load
            int w  = wo0 + i + dw;
            int wc = w < 0 ? 0 : (w > W_ - 1 ? W_ - 1 : w);
            vm[i] = ((w >= 0 && w < W_) ? 1.f : 0.f) * rowv;
            g[i]  = rp[wc];
        }
        v8h pack;
        #pragma unroll
        for (int i = 0; i < 8; ++i) pack[i] = (_Float16)(g[i] * vm[i]);
        *(v8h*)&b_tile[(size_t)buf * KC * BST + bkrow * BST + bpix0] = pack;
    };

    v8f acc = {};
    build(0, 0);
    __syncthreads();

    for (int ch = 0; ch < NCHUNK; ++ch) {
        const int cur = ch & 1;
        const _Float16* bt = &b_tile[(size_t)cur * KC * BST];

        Frag a, bf;
        const _Float16* ap = &a_lds[(size_t)mrow * CK + ch * KC + khalf * 8];
        a.h[0] = *(const v8h*)ap;
        a.h[1] = *(const v8h*)(ap + 16);
        const _Float16* bp = &bt[(khalf * 8) * BST + ntile * 16 + (lane & 15)];
        #pragma unroll
        for (int s = 0; s < 8; ++s) { bf.v[s] = bp[s * BST]; bf.v[8 + s] = bp[(16 + s) * BST]; }
        acc = __builtin_amdgcn_wmma_f32_16x16x32_f16(false, a.v, false, bf.v,
                                                     (short)0, acc, false, false);
        if (ch + 1 < NCHUNK) build(ch + 1, cur ^ 1);   // gathers overlap the WMMA
        __syncthreads();
    }

    // epilogue: rows 0-17 -> offset, 18-26 -> mask, 27-31 dropped
    const int mbase = mtile * 16 + (lane >> 4) * 8;
    const int ncol  = pixbase + ntile * 16 + (lane & 15);
    #pragma unroll
    for (int v = 0; v < 8; ++v) {
        int m = mbase + v;
        if (m < OFF_CH)
            offset[((size_t)b * OFF_CH + m) * HW + ncol] = acc[v] + b_off[m];
        else if (m < 27)
            mask[((size_t)b * MOD_CH + (m - OFF_CH)) * HW + ncol] = acc[v] + b_mod[m - OFF_CH];
    }
}

// ===========================================================================
// Kernel 2: fused bilinear sampling + WMMA GEMM (M=64)
// ===========================================================================
struct SInfo { int i00, i01, i10, i11; float w00, w01, w10, w11; };

constexpr size_t K2_A_BYTES  = (size_t)CO_ * CK * 2;              // 73,728
constexpr size_t K2_BT_BYTES = (size_t)2 * KC * BST * 2;          // 17,408
constexpr size_t K2_SI_BYTES = (size_t)KPOS * NT * sizeof(SInfo); // 36,864
constexpr size_t K2_SMEM     = K2_A_BYTES + K2_BT_BYTES + K2_SI_BYTES; // 128,000

__global__ void __launch_bounds__(NTHREADS)
deform_wmma(const float* __restrict__ x,
            const float* __restrict__ offset, const float* __restrict__ mask,
            const float* __restrict__ w_d, const float* __restrict__ b_d,
            float* __restrict__ out)
{
    extern __shared__ __align__(16) char smem[];
    _Float16* a_lds  = (_Float16*)smem;                          // [64][576] f16
    _Float16* b_tile = (_Float16*)(smem + K2_A_BYTES);           // [2][32][BST] f16
    SInfo*    sinfo  = (SInfo*)(smem + K2_A_BYTES + K2_BT_BYTES);// [9][128]

    const int tid = threadIdx.x;
    const int bid = blockIdx.x;
    const int b       = bid >> 7;
    const int pixbase = (bid & 127) * NT;

    // A = w_d as f16 (float4 -> packed 4xf16)
    for (int i = tid; i < CO_ * CK / 4; i += NTHREADS) {
        float4 v = *(const float4*)&w_d[i * 4];
        v4h h4; h4[0] = (_Float16)v.x; h4[1] = (_Float16)v.y;
                h4[2] = (_Float16)v.z; h4[3] = (_Float16)v.w;
        *(v4h*)&a_lds[i * 4] = h4;
    }

    // per-(k,pixel) bilinear metadata; mask & per-corner validity folded in.
    for (int p = tid; p < KPOS * NT; p += NTHREADS) {
        int kpos = p >> 7, pix = p & 127;
        int n  = pixbase + pix;
        int ho = n >> 7, wo = n & 127;
        float dy = __builtin_nontemporal_load(&offset[((size_t)b * OFF_CH + (kpos * 2 + 0)) * HW + n]);
        float dx = __builtin_nontemporal_load(&offset[((size_t)b * OFF_CH + (kpos * 2 + 1)) * HW + n]);
        float mv = __builtin_nontemporal_load(&mask  [((size_t)b * MOD_CH + kpos) * HW + n]);
        float y  = (float)(ho - 1 + kpos / 3) + dy;
        float xs = (float)(wo - 1 + kpos % 3) + dx;
        float y0f = floorf(y), x0f = floorf(xs);
        int y0 = (int)y0f, x0 = (int)x0f, y1 = y0 + 1, x1 = x0 + 1;
        float wy1 = y - y0f, wx1 = xs - x0f;
        float wy0 = 1.f - wy1, wx0 = 1.f - wx1;
        float vy0 = (y0 >= 0 && y0 < H_) ? 1.f : 0.f;
        float vy1 = (y1 >= 0 && y1 < H_) ? 1.f : 0.f;
        float vx0 = (x0 >= 0 && x0 < W_) ? 1.f : 0.f;
        float vx1 = (x1 >= 0 && x1 < W_) ? 1.f : 0.f;
        int yc0 = y0 < 0 ? 0 : (y0 > H_ - 1 ? H_ - 1 : y0);
        int yc1 = y1 < 0 ? 0 : (y1 > H_ - 1 ? H_ - 1 : y1);
        int xc0 = x0 < 0 ? 0 : (x0 > W_ - 1 ? W_ - 1 : x0);
        int xc1 = x1 < 0 ? 0 : (x1 > W_ - 1 ? W_ - 1 : x1);
        SInfo s;
        s.i00 = yc0 * W_ + xc0;  s.i01 = yc0 * W_ + xc1;
        s.i10 = yc1 * W_ + xc0;  s.i11 = yc1 * W_ + xc1;
        s.w00 = wy0 * wx0 * mv * vy0 * vx0;
        s.w01 = wy0 * wx1 * mv * vy0 * vx1;
        s.w10 = wy1 * wx0 * mv * vy1 * vx0;
        s.w11 = wy1 * wx1 * mv * vy1 * vx1;
        sinfo[p] = s;
    }

    const float* xb = x + (size_t)b * C_ * HW;
    const int lane  = tid & 31;
    const int wave  = tid >> 5;                  // 16 waves
    const int mtile = wave & 3;                  // 4 M-tiles
    const int npair = wave >> 2;                 // 4 N-tile pairs (8 N-tiles)
    const int mrow  = mtile * 16 + (lane & 15);
    const int khalf = lane >> 4;

    const int bkrow = (tid * 8) >> 7;
    const int bpix0 = (tid * 8) & 127;

    auto build = [&](int ch, int buf) {
        int kt = ch * KC + bkrow;
        int c  = kt / KPOS, kp = kt - c * KPOS;
        const float* xc = xb + (size_t)c * HW;
        const SInfo* sp = &sinfo[kp * NT + bpix0];
        SInfo s[8];
        #pragma unroll
        for (int i = 0; i < 8; ++i) s[i] = sp[i];
        // issue all 32 gathers before any use -> full MLP under the WMMAs
        float g00[8], g01[8], g10[8], g11[8];
        #pragma unroll
        for (int i = 0; i < 8; ++i) {
            g00[i] = xc[s[i].i00];  g01[i] = xc[s[i].i01];
            g10[i] = xc[s[i].i10];  g11[i] = xc[s[i].i11];
        }
        v8h pack;
        #pragma unroll
        for (int i = 0; i < 8; ++i)
            pack[i] = (_Float16)(g00[i] * s[i].w00 + g01[i] * s[i].w01 +
                                 g10[i] * s[i].w10 + g11[i] * s[i].w11);
        *(v8h*)&b_tile[(size_t)buf * KC * BST + bkrow * BST + bpix0] = pack;
    };

    v8f acc0 = {}; v8f acc1 = {};
    __syncthreads();                              // sinfo ready
    build(0, 0);
    __syncthreads();

    for (int ch = 0; ch < NCHUNK; ++ch) {
        const int cur = ch & 1;
        const _Float16* bt = &b_tile[(size_t)cur * KC * BST];

        Frag a, b0, b1;
        const _Float16* ap = &a_lds[(size_t)mrow * CK + ch * KC + khalf * 8];
        a.h[0] = *(const v8h*)ap;
        a.h[1] = *(const v8h*)(ap + 16);
        const _Float16* bp = &bt[(khalf * 8) * BST + (npair * 2 + 0) * 16 + (lane & 15)];
        const _Float16* bq = &bt[(khalf * 8) * BST + (npair * 2 + 1) * 16 + (lane & 15)];
        #pragma unroll
        for (int s = 0; s < 8; ++s) {
            b0.v[s] = bp[s * BST]; b0.v[8 + s] = bp[(16 + s) * BST];
            b1.v[s] = bq[s * BST]; b1.v[8 + s] = bq[(16 + s) * BST];
        }
        acc0 = __builtin_amdgcn_wmma_f32_16x16x32_f16(false, a.v, false, b0.v,
                                                      (short)0, acc0, false, false);
        acc1 = __builtin_amdgcn_wmma_f32_16x16x32_f16(false, a.v, false, b1.v,
                                                      (short)0, acc1, false, false);
        if (ch + 1 < NCHUNK) {
            build(ch + 1, cur ^ 1);               // gathers overlap the WMMAs
            if (lane == 0) {                      // warm next chunk's channel planes
                int cn = ((ch + 1) * KC) / KPOS;
                __builtin_prefetch(xb + (size_t)cn * HW + wave * 512, 0, 1);
            }
        }
        __syncthreads();
    }

    // epilogue: C/D layout (M = v + 8*(lane>=16), N = lane%16); streamed out
    const int mbase = mtile * 16 + (lane >> 4) * 8;
    const int ncol  = lane & 15;
    const int n0    = pixbase + (npair * 2 + 0) * 16 + ncol;
    const int n1    = pixbase + (npair * 2 + 1) * 16 + ncol;
    #pragma unroll
    for (int v = 0; v < 8; ++v) {
        int m = mbase + v;
        float bias = b_d[m];
        __builtin_nontemporal_store(acc0[v] + bias, &out[((size_t)b * CO_ + m) * HW + n0]);
        __builtin_nontemporal_store(acc1[v] + bias, &out[((size_t)b * CO_ + m) * HW + n1]);
    }
}

// =============================== launcher ==================================
extern "C" void kernel_launch(void* const* d_in, const int* in_sizes, int n_in,
                              void* d_out, int out_size, void* d_ws, size_t ws_size,
                              hipStream_t stream)
{
    (void)in_sizes; (void)n_in; (void)out_size; (void)ws_size;
    const float* x     = (const float*)d_in[0];
    const float* w_off = (const float*)d_in[1];
    const float* b_off = (const float*)d_in[2];
    const float* w_mod = (const float*)d_in[3];
    const float* b_mod = (const float*)d_in[4];
    const float* w_d   = (const float*)d_in[5];
    const float* b_d   = (const float*)d_in[6];
    float* out = (float*)d_out;

    // workspace: offset (B*18*HW) + mask (B*9*HW) fp32 => ~6.75 MB (lives in L2)
    float* offset = (float*)d_ws;
    float* mask   = offset + (size_t)B_ * OFF_CH * HW;

    const int blocks = B_ * (HW / NT);            // 512

    conv_offmask_wmma<<<blocks, NTHREADS, K1_SMEM, stream>>>(
        x, w_off, b_off, w_mod, b_mod, offset, mask);

    deform_wmma<<<blocks, NTHREADS, K2_SMEM, stream>>>(
        x, offset, mask, w_d, b_d, out);
}
